// TextRNN_87514253623657
// MI455X (gfx1250) — matrix operand order repaired and checked by reading, not compiled, not run on hardware
//
#include <hip/hip_runtime.h>
#include <hip/hip_bf16.h>
#include <math.h>

// ---------------------------------------------------------------------------
// TextRNN (embedding -> BiLSTM(64) -> BiLSTM(32) -> maxpool -> BN -> dense ->
// softmax) for MI455X / gfx1250.  Matrix work runs on v_wmma_f32_16x16x32_f16.
// GEMMs block 4 n-tiles x 2 directions per wave (8 independent WMMA chains,
// 8x A-fragment reuse); the serial LSTM loops prefetch next-step Z rows.
// ---------------------------------------------------------------------------

typedef __attribute__((ext_vector_type(16))) _Float16 v16h;
typedef __attribute__((ext_vector_type(8)))  float    v8f;

static constexpr int B_    = 256;
static constexpr int T_    = 256;
static constexpr int E_    = 300;
static constexpr int EPAD  = 320;   // pad K to multiple of 32 for WMMA
static constexpr int U1_   = 64;
static constexpr int NG1   = 4 * U1_;   // 256
static constexpr int U2_   = 32;
static constexpr int NG2   = 4 * U2_;   // 128
static constexpr int C_    = 5;
static constexpr float BN_EPS = 1e-3f;

__device__ __forceinline__ float sigf(float x) { return 1.0f / (1.0f + expf(-x)); }

// --- WMMA fragment packing helpers (ISA 7.12.2 layouts, wave32) -------------
// A (16x32 f16):  lane = (m&15) + 16*((kk>>3)&1),  slot = (kk&7) + 8*(kk>=16)
// B (32x16 f16):  lane = (n&15) + 16*((kk>>3)&1),  slot = (kk&7) + 8*(kk>=16)
// C/D (16x16 f32): row = 16*mt + r + 8*(lane>=16), col = 16*nt + (lane&15)

// Pack a row-major f32 weight [K,N] (K padded to Kpad) into B fragments.
__global__ void pack_w_kernel(const float* __restrict__ W, _Float16* __restrict__ out,
                              int K, int Kpad, int N) {
    int idx = blockIdx.x * blockDim.x + threadIdx.x;
    if (idx >= Kpad * N) return;
    int k = idx / N, n = idx % N;
    int kt = k >> 5, kk = k & 31, nt = n >> 4;
    int lane = (n & 15) + (((kk >> 3) & 1) << 4);
    int s = (kk & 7) + ((kk & 16) ? 8 : 0);
    int NT = N >> 4;
    float v = (k < K) ? W[(size_t)k * N + n] : 0.0f;
    out[(((size_t)kt * NT + nt) * 32 + lane) * 16 + s] = (_Float16)v;
}

// Embedding gather -> f16, written directly as A fragments [M=B*T, Kpad=320].
__global__ void embed_pack_kernel(const int* __restrict__ tokens,
                                  const float* __restrict__ emb,
                                  _Float16* __restrict__ Xpk) {
    size_t idx = (size_t)blockIdx.x * blockDim.x + threadIdx.x;
    const size_t total = (size_t)B_ * T_ * EPAD;
    if (idx >= total) return;
    int m = (int)(idx / EPAD);
    int k = (int)(idx % EPAD);
    float v = 0.0f;
    if (k < E_) {
        int tok = tokens[m];
        v = emb[(size_t)tok * E_ + k];
    }
    int mt = m >> 4, kt = k >> 5, kk = k & 31;
    int lane = (m & 15) + (((kk >> 3) & 1) << 4);
    int s = (kk & 7) + ((kk & 16) ? 8 : 0);
    Xpk[(((size_t)mt * (EPAD / 32) + kt) * 32 + lane) * 16 + s] = (_Float16)v;
}

// Dual-direction GEMM:  C{0,1}[M,N](f16) = Apk @ Bpk{0,1} + bias{0,1}.
// Each wave: one m-tile x 4 consecutive n-tiles x 2 weight sets.
// A fragment is loaded once per k-step and feeds 8 independent WMMA chains.
__global__ void gemm_wmma_f16_dual(const _Float16* __restrict__ Apk,
                                   const _Float16* __restrict__ Bpk0,
                                   const _Float16* __restrict__ Bpk1,
                                   const float* __restrict__ bias0,
                                   const float* __restrict__ bias1,
                                   _Float16* __restrict__ C0,
                                   _Float16* __restrict__ C1,
                                   int MT, int NT, int KT, int N) {
    constexpr int NB = 4;                 // n-tiles per wave
    int wv = threadIdx.x >> 5;
    int lane = threadIdx.x & 31;
    int g = blockIdx.x * 8 + wv;
    int NGRP = NT / NB;
    if (g >= MT * NGRP) return;           // wave-uniform guard
    int mt = g / NGRP;
    int nt0 = (g % NGRP) * NB;

    v8f acc[2][NB];
#pragma unroll
    for (int d = 0; d < 2; ++d)
#pragma unroll
        for (int i = 0; i < NB; ++i) acc[d][i] = (v8f){};

    for (int kt = 0; kt < KT; ++kt) {
        v16h a = *(const v16h*)(Apk + (((size_t)mt * KT + kt) * 32 + lane) * 16);
        v16h b0[NB], b1[NB];
#pragma unroll
        for (int i = 0; i < NB; ++i) {
            size_t boff = (((size_t)kt * NT + nt0 + i) * 32 + lane) * 16;
            b0[i] = *(const v16h*)(Bpk0 + boff);
            b1[i] = *(const v16h*)(Bpk1 + boff);
        }
#pragma unroll
        for (int i = 0; i < NB; ++i) {
            acc[0][i] = __builtin_amdgcn_wmma_f32_16x16x32_f16(
                false, a, false, b0[i], (short)0, acc[0][i], false, false);
            acc[1][i] = __builtin_amdgcn_wmma_f32_16x16x32_f16(
                false, a, false, b1[i], (short)0, acc[1][i], false, false);
        }
    }

    int lnlo = lane & 15, hi = lane >> 4;
    int rowb = mt * 16 + hi * 8;
#pragma unroll
    for (int i = 0; i < NB; ++i) {
        int col = (nt0 + i) * 16 + lnlo;
        float bv0 = bias0[col], bv1 = bias1[col];
#pragma unroll
        for (int r = 0; r < 8; ++r) {
            C0[(size_t)(rowb + r) * N + col] = (_Float16)(acc[0][i][r] + bv0);
            C1[(size_t)(rowb + r) * N + col] = (_Float16)(acc[1][i][r] + bv1);
        }
    }
}

// Persistent recurrent LSTM layer.  grid.x = 4 batch-slices * 2 directions.
// Per step: G = h @ Wr via WMMA (h packed in LDS), then fused gate math with
// cell state in registers.  OUTMODE 0: scatter h into layer-2 A-fragment
// buffer (fwd cols 0..63, bwd 64..127).  OUTMODE 1: write f32 H2[B*T,64].
template <int U, int OUTMODE>
__global__ void lstm_rec_kernel(const _Float16* __restrict__ Zf,
                                const _Float16* __restrict__ Zb,
                                const _Float16* __restrict__ Wrf,
                                const _Float16* __restrict__ Wrb,
                                void* __restrict__ outBuf) {
    constexpr int NG  = 4 * U;        // gate width
    constexpr int KT  = U / 32;       // k-tiles of recurrent GEMM (2 or 1)
    constexpr int NT  = NG / 16;      // n-tiles (16 or 8)
    constexpr int MT  = 4;            // 64 batch rows per block
    constexpr int NTW = NT / 8;       // n-tiles per wave (2 or 1)
    constexpr int NC  = 64 * U / 256; // (b,u) cells per thread (16 or 8)
    constexpr int CH  = NG / 64;      // 128B cache lines per Z row (4 or 2)

    extern __shared__ __align__(128) char smem[];
    float*    G   = (float*)smem;                                // [64][NG]
    _Float16* hpk = (_Float16*)(smem + 64 * NG * sizeof(float)); // packed A frags

    const int tid  = threadIdx.x;
    const int lane = tid & 31;
    const int wv   = tid >> 5;
    const int dir   = blockIdx.x & 1;
    const int slice = blockIdx.x >> 1;
    const int bbase = slice * 64;

    const _Float16* Zin = dir ? Zb : Zf;
    const _Float16* Wr  = dir ? Wrb : Wrf;

    // h0 = 0
    for (int i = tid; i < MT * KT * 32 * 16; i += 256) hpk[i] = (_Float16)0.0f;

    // Preload recurrent-weight B fragments for this wave's n-tiles.
    v16h bfr[NTW][KT];
#pragma unroll
    for (int ni = 0; ni < NTW; ++ni)
#pragma unroll
        for (int kt = 0; kt < KT; ++kt) {
            int nt = wv * NTW + ni;
            bfr[ni][kt] = *(const v16h*)(Wr + (((size_t)kt * NT + nt) * 32 + lane) * 16);
        }

    float creg[NC];
#pragma unroll
    for (int j = 0; j < NC; ++j) creg[j] = 0.0f;

    // Prefetch step-0 Z rows while waves warm up.
    {
        int lmP = tid / CH, chP = tid % CH;
        if (lmP < 64) {
            int t0 = dir ? (T_ - 1) : 0;
            __builtin_prefetch(Zin + (size_t)((bbase + lmP) * T_ + t0) * NG + chP * 64,
                               0, 1);
        }
    }

    for (int step = 0; step < T_; ++step) {
        __syncthreads();   // hpk ready (init or previous gate phase); G free

        // ---- recurrent GEMM: G[64,NG] = h[64,U] @ Wr[U,NG] ----
        v16h afr[MT][KT];
#pragma unroll
        for (int mt = 0; mt < MT; ++mt)
#pragma unroll
            for (int kt = 0; kt < KT; ++kt)
                afr[mt][kt] = *(const v16h*)(hpk + ((mt * KT + kt) * 32 + lane) * 16);

        // Prefetch next step's Z rows; overlaps the WMMA phase below.
        if (step + 1 < T_) {
            int lmP = tid / CH, chP = tid % CH;
            if (lmP < 64) {
                int tn = dir ? (T_ - 2 - step) : (step + 1);
                __builtin_prefetch(
                    Zin + (size_t)((bbase + lmP) * T_ + tn) * NG + chP * 64, 0, 1);
            }
        }

#pragma unroll
        for (int ni = 0; ni < NTW; ++ni) {
            int nt = wv * NTW + ni;
#pragma unroll
            for (int mt = 0; mt < MT; ++mt) {
                v8f acc = {};
#pragma unroll
                for (int kt = 0; kt < KT; ++kt)
                    acc = __builtin_amdgcn_wmma_f32_16x16x32_f16(
                        false, afr[mt][kt], false, bfr[ni][kt],
                        (short)0, acc, false, false);
                int lnlo = lane & 15, hi = lane >> 4;
                int col = nt * 16 + lnlo;
                int rowb = mt * 16 + hi * 8;
#pragma unroll
                for (int r = 0; r < 8; ++r)
                    G[(rowb + r) * NG + col] = acc[r];
            }
        }
        __syncthreads();   // G complete

        // ---- gate math: z = Zin + G; c,h update; repack h ----
        const int t = dir ? (T_ - 1 - step) : step;
#pragma unroll
        for (int j = 0; j < NC; ++j) {
            int c0 = tid + j * 256;
            int lm = c0 / U;
            int u  = c0 % U;
            int b  = bbase + lm;
            size_t zrow = (size_t)(b * T_ + t) * NG;
            float zi = (float)Zin[zrow + u]         + G[lm * NG + u];
            float zf = (float)Zin[zrow + u + U]     + G[lm * NG + u + U];
            float zg = (float)Zin[zrow + u + 2 * U] + G[lm * NG + u + 2 * U];
            float zo = (float)Zin[zrow + u + 3 * U] + G[lm * NG + u + 3 * U];
            float cn = sigf(zf) * creg[j] + sigf(zi) * tanhf(zg);
            float h  = sigf(zo) * tanhf(cn);
            creg[j] = cn;

            // repack h into LDS A-fragment layout for next step
            int kk = u & 31, kt = u >> 5;
            int pl = (lm & 15) + (((kk >> 3) & 1) << 4);
            int ps = (kk & 7) + ((kk & 16) ? 8 : 0);
            int pmt = lm >> 4;
            hpk[((pmt * KT + kt) * 32 + pl) * 16 + ps] = (_Float16)h;

            if (OUTMODE == 0) {
                // scatter into layer-2 input A fragments: K2=128, KT2=4
                int m2 = b * T_ + t;
                int k2 = u + 64 * dir;
                int kk2 = k2 & 31, kt2 = k2 >> 5;
                int l2 = (m2 & 15) + (((kk2 >> 3) & 1) << 4);
                int s2 = (kk2 & 7) + ((kk2 & 16) ? 8 : 0);
                int mt2 = m2 >> 4;
                ((_Float16*)outBuf)[(((size_t)mt2 * 4 + kt2) * 32 + l2) * 16 + s2] =
                    (_Float16)h;
            } else {
                ((float*)outBuf)[(size_t)(b * T_ + t) * 64 + u + 32 * dir] = h;
            }
        }
    }
}

// max over T -> BatchNorm -> dense(64->5) -> softmax.  One block per batch row.
__global__ void head_kernel(const float* __restrict__ H2,
                            const float* __restrict__ gamma,
                            const float* __restrict__ beta,
                            const float* __restrict__ mean,
                            const float* __restrict__ var,
                            const float* __restrict__ dw,
                            const float* __restrict__ db,
                            float* __restrict__ out) {
    __shared__ float y[64];
    __shared__ float logits[C_];
    int b = blockIdx.x;
    int f = threadIdx.x;   // 64 threads
    float m = -3.4e38f;
    for (int t = 0; t < T_; ++t)
        m = fmaxf(m, H2[(size_t)(b * T_ + t) * 64 + f]);
    y[f] = gamma[f] * (m - mean[f]) * rsqrtf(var[f] + BN_EPS) + beta[f];
    __syncthreads();
    if (f < C_) {
        float l = db[f];
        for (int k = 0; k < 64; ++k) l += y[k] * dw[k * C_ + f];
        logits[f] = l;
    }
    __syncthreads();
    if (f == 0) {
        float mx = logits[0];
        for (int c = 1; c < C_; ++c) mx = fmaxf(mx, logits[c]);
        float e[C_], s = 0.0f;
        for (int c = 0; c < C_; ++c) { e[c] = expf(logits[c] - mx); s += e[c]; }
        for (int c = 0; c < C_; ++c) out[b * C_ + c] = e[c] / s;
    }
}

extern "C" void kernel_launch(void* const* d_in, const int* in_sizes, int n_in,
                              void* d_out, int out_size, void* d_ws, size_t ws_size,
                              hipStream_t stream) {
    (void)in_sizes; (void)n_in; (void)out_size; (void)ws_size;

    const int*   tokens = (const int*)d_in[0];
    const float* emb    = (const float*)d_in[1];
    const float* wk1f = (const float*)d_in[2];
    const float* wr1f = (const float*)d_in[3];
    const float* b1f  = (const float*)d_in[4];
    const float* wk1b = (const float*)d_in[5];
    const float* wr1b = (const float*)d_in[6];
    const float* b1b  = (const float*)d_in[7];
    const float* wk2f = (const float*)d_in[8];
    const float* wr2f = (const float*)d_in[9];
    const float* b2f  = (const float*)d_in[10];
    const float* wk2b = (const float*)d_in[11];
    const float* wr2b = (const float*)d_in[12];
    const float* b2b  = (const float*)d_in[13];
    const float* bn_g = (const float*)d_in[14];
    const float* bn_b = (const float*)d_in[15];
    const float* bn_m = (const float*)d_in[16];
    const float* bn_v = (const float*)d_in[17];
    const float* dw   = (const float*)d_in[18];
    const float* db   = (const float*)d_in[19];

    // ---- carve workspace (256B aligned) ----
    char* p = (char*)d_ws;
    auto carve = [&p](size_t bytes) -> char* {
        char* r = p;
        p += (bytes + 255) & ~((size_t)255);
        return r;
    };
    const size_t M = (size_t)B_ * T_;   // 65536 rows

    _Float16* X1pk   = (_Float16*)carve(M * EPAD * 2);
    _Float16* Wk1fPk = (_Float16*)carve((size_t)EPAD * NG1 * 2);
    _Float16* Wk1bPk = (_Float16*)carve((size_t)EPAD * NG1 * 2);
    _Float16* Wr1fPk = (_Float16*)carve((size_t)U1_ * NG1 * 2);
    _Float16* Wr1bPk = (_Float16*)carve((size_t)U1_ * NG1 * 2);
    _Float16* Wk2fPk = (_Float16*)carve((size_t)(2 * U1_) * NG2 * 2);
    _Float16* Wk2bPk = (_Float16*)carve((size_t)(2 * U1_) * NG2 * 2);
    _Float16* Wr2fPk = (_Float16*)carve((size_t)U2_ * NG2 * 2);
    _Float16* Wr2bPk = (_Float16*)carve((size_t)U2_ * NG2 * 2);
    _Float16* Z1f    = (_Float16*)carve(M * NG1 * 2);
    _Float16* Z1b    = (_Float16*)carve(M * NG1 * 2);
    _Float16* X2pk   = (_Float16*)carve(M * 128 * 2);
    _Float16* Z2f    = (_Float16*)carve(M * NG2 * 2);
    _Float16* Z2b    = (_Float16*)carve(M * NG2 * 2);
    float*    H2     = (float*)carve(M * 64 * 4);

    // 1) embedding gather -> packed A fragments
    {
        size_t total = M * EPAD;
        embed_pack_kernel<<<(unsigned)((total + 255) / 256), 256, 0, stream>>>(
            tokens, emb, X1pk);
    }

    // 2) pack all weights into B fragments
    auto packW = [&](const float* W, _Float16* o, int K, int Kpad, int N) {
        int total = Kpad * N;
        pack_w_kernel<<<(total + 255) / 256, 256, 0, stream>>>(W, o, K, Kpad, N);
    };
    packW(wk1f, Wk1fPk, E_, EPAD, NG1);
    packW(wk1b, Wk1bPk, E_, EPAD, NG1);
    packW(wr1f, Wr1fPk, U1_, U1_, NG1);
    packW(wr1b, Wr1bPk, U1_, U1_, NG1);
    packW(wk2f, Wk2fPk, 2 * U1_, 2 * U1_, NG2);
    packW(wk2b, Wk2bPk, 2 * U1_, 2 * U1_, NG2);
    packW(wr2f, Wr2fPk, U2_, U2_, NG2);
    packW(wr2b, Wr2bPk, U2_, U2_, NG2);

    // 3) layer-1 input projections for BOTH directions in one pass:
    //    [65536,320]@[320,256]+b  with shared A fragments
    {
        int MT = (int)(M / 16), NT = NG1 / 16, KT = EPAD / 32;   // 4096,16,10
        int groups = MT * (NT / 4);
        gemm_wmma_f16_dual<<<(groups + 7) / 8, 256, 0, stream>>>(
            X1pk, Wk1fPk, Wk1bPk, b1f, b1b, Z1f, Z1b, MT, NT, KT, NG1);
    }

    // 4) BiLSTM-1 recurrence (4 batch slices x 2 directions), h -> X2pk
    {
        size_t shmem = 64 * NG1 * sizeof(float) + 64 * U1_ * sizeof(_Float16); // 72KB
        lstm_rec_kernel<U1_, 0><<<8, 256, (unsigned)shmem, stream>>>(
            Z1f, Z1b, Wr1fPk, Wr1bPk, (void*)X2pk);
    }

    // 5) layer-2 input projections (both directions): [65536,128]@[128,128]+b
    {
        int MT = (int)(M / 16), NT = NG2 / 16, KT = (2 * U1_) / 32;  // 4096,8,4
        int groups = MT * (NT / 4);
        gemm_wmma_f16_dual<<<(groups + 7) / 8, 256, 0, stream>>>(
            X2pk, Wk2fPk, Wk2bPk, b2f, b2b, Z2f, Z2b, MT, NT, KT, NG2);
    }

    // 6) BiLSTM-2 recurrence, h -> H2 (f32 [B*T,64], fwd|bwd concat)
    {
        size_t shmem = 64 * NG2 * sizeof(float) + 64 * U2_ * sizeof(_Float16); // 36KB
        lstm_rec_kernel<U2_, 1><<<8, 256, (unsigned)shmem, stream>>>(
            Z2f, Z2b, Wr2fPk, Wr2bPk, (void*)H2);
    }

    // 7) maxpool over T -> BN -> dense -> softmax
    head_kernel<<<B_, 64, 0, stream>>>(H2, bn_g, bn_b, bn_m, bn_v, dw, db,
                                       (float*)d_out);
}
